// BertLayer_47339129536519
// MI455X (gfx1250) — matrix operand order repaired and checked by reading, not compiled, not run on hardware
//
#include <hip/hip_runtime.h>
#include <hip/hip_bf16.h>

typedef __attribute__((ext_vector_type(16))) _Float16 v16h;
typedef __attribute__((ext_vector_type(8)))  _Float16 v8h;
typedef __attribute__((ext_vector_type(8)))  float    v8f;

#define HIDN 768
#define NHEAD 12
#define HDIM 64
#define SEQ 512
#define BATCH 32
#define NTOK (BATCH * SEQ)   // 16384
#define FFN (4 * HIDN)       // 3072
#define LDSL (128 * 40)      // one LDS tile buffer (halves)

// ---------------------------------------------------------------------------
// elementwise f32 -> f16
__global__ void cvt_f32_f16_kernel(const float* __restrict__ in,
                                   _Float16* __restrict__ out, int n) {
  int i = blockIdx.x * blockDim.x + threadIdx.x;
  if (i < n) out[i] = (_Float16)in[i];
}

// weight transpose: W[K][N] f32  ->  Bt[N][K] f16
__global__ void transpose_w_kernel(const float* __restrict__ W,
                                   _Float16* __restrict__ Bt, int K, int N) {
  int i = blockIdx.x * blockDim.x + threadIdx.x;
  if (i >= K * N) return;
  int n = i / K, k = i - n * K;
  Bt[i] = (_Float16)W[(size_t)k * N + n];
}

// V [token][hid] f16 -> Vt [b][h][d][s] f16
__global__ void transpose_v_kernel(const _Float16* __restrict__ V,
                                   _Float16* __restrict__ Vt) {
  int i = blockIdx.x * blockDim.x + threadIdx.x;
  if (i >= NTOK * HIDN) return;
  int s = i % SEQ;
  int d = (i / SEQ) % HDIM;
  int h = (i / (SEQ * HDIM)) % NHEAD;
  int b = i / (SEQ * HDIM * NHEAD);
  Vt[i] = V[((size_t)(b * SEQ + s)) * HIDN + h * HDIM + d];
}

// gather CLS rows (token 0 of each of the 32 sequences)
__global__ void gather_cls_kernel(const float* __restrict__ x1,
                                  const _Float16* __restrict__ x1h,
                                  float* __restrict__ clsf,
                                  _Float16* __restrict__ clsh) {
  int i = blockIdx.x * blockDim.x + threadIdx.x;
  if (i >= 32 * HIDN) return;
  int b = i / HIDN, c = i - b * HIDN;
  size_t src = (size_t)b * SEQ * HIDN + c;
  clsf[i] = x1[src];
  clsh[i] = x1h[src];
}

__global__ void scatter_cls_kernel(const float* __restrict__ dx1,
                                   const _Float16* __restrict__ dx1h,
                                   float* __restrict__ x1,
                                   _Float16* __restrict__ x1h) {
  int i = blockIdx.x * blockDim.x + threadIdx.x;
  if (i >= 32 * HIDN) return;
  int b = i / HIDN, c = i - b * HIDN;
  size_t dst = (size_t)b * SEQ * HIDN + c;
  x1[dst] = dx1[i];
  x1h[dst] = dx1h[i];
}

// ---------------------------------------------------------------------------
// LayerNorm over width=768 (256 threads, 3 elems/thread); optional f16 copy
__global__ __launch_bounds__(256)
void ln_kernel(const float* __restrict__ in, const float* __restrict__ g,
               const float* __restrict__ bb, float* __restrict__ out32,
               _Float16* __restrict__ out16) {
  int row = blockIdx.x;
  int tid = threadIdx.x;
  const float* x = in + (size_t)row * HIDN;
  float loc[3];
  float s = 0.f;
#pragma unroll
  for (int i = 0; i < 3; ++i) { loc[i] = x[tid + i * 256]; s += loc[i]; }
  __shared__ float red[256];
  red[tid] = s; __syncthreads();
  for (int off = 128; off > 0; off >>= 1) {
    if (tid < off) red[tid] += red[tid + off];
    __syncthreads();
  }
  float mean = red[0] * (1.0f / HIDN);
  __syncthreads();
  float v = 0.f;
#pragma unroll
  for (int i = 0; i < 3; ++i) { float d = loc[i] - mean; v += d * d; }
  red[tid] = v; __syncthreads();
  for (int off = 128; off > 0; off >>= 1) {
    if (tid < off) red[tid] += red[tid + off];
    __syncthreads();
  }
  float rstd = rsqrtf(red[0] * (1.0f / HIDN) + 1e-12f);
#pragma unroll
  for (int i = 0; i < 3; ++i) {
    int c = tid + i * 256;
    float y = (loc[i] - mean) * rstd * g[c] + bb[c];
    out32[(size_t)row * HIDN + c] = y;
    if (out16) out16[(size_t)row * HIDN + c] = (_Float16)y;
  }
}

// ---------------------------------------------------------------------------
// WMMA GEMM body: C[M][N] = A_f16[M][K] @ Bt_f16[N][K]^T + bias (+res) (gelu?)
// 256 thr (8 waves), 128x128 tile, K-step 32, ping-pong double-buffered LDS.
// GUARD=false fast path: async global->LDS DMA staging (ASYNCcnt), no VGPR
// round-trip.  GUARD=true (M not multiple of 128): VGPR staging with zero-fill.
template <bool GUARD>
__device__ __forceinline__ void gemm_body(
    const _Float16* __restrict__ A, const _Float16* __restrict__ Bt,
    const float* __restrict__ bias, const float* __restrict__ res,
    float* __restrict__ C32, _Float16* __restrict__ C16,
    int M, int N, int K, int do_gelu, _Float16* As, _Float16* Bs) {
  int tid = threadIdx.x;
  int lane = tid & 31, wid = tid >> 5;
  int waveM = wid >> 2, waveN = wid & 3;
  int l15 = lane & 15, hs = lane >> 4;
  size_t blockM = (size_t)blockIdx.y * 128;
  size_t blockN = (size_t)blockIdx.x * 128;

  v8f z = {};
  v8f acc[4][2];
#pragma unroll
  for (int i = 0; i < 4; ++i)
#pragma unroll
    for (int j = 0; j < 2; ++j) acc[i][j] = z;

  int ldRow = tid >> 1;
  int ldCol = (tid & 1) * 16;
  int sBase = ldRow * 40 + ldCol;
  bool rowOK = true;
  if (GUARD) rowOK = (blockM + ldRow) < (size_t)M;
  const _Float16* aP = A + (blockM + ldRow) * K + ldCol;
  const _Float16* bP = Bt + (blockN + ldRow) * K + ldCol;

  v8h hz = {};
  // prologue: stage K-tile 0 into buffer 0
  if (!GUARD) {
    unsigned la = (unsigned)(size_t)&As[sBase];
    unsigned lb = (unsigned)(size_t)&Bs[sBase];
    // each thread DMAs 2x16B straight into LDS (offset applies to both sides)
    asm volatile("global_load_async_to_lds_b128 %0, %1, off\n\t"
                 "global_load_async_to_lds_b128 %0, %1, off offset:16"
                 :: "v"(la), "v"(aP) : "memory");
    asm volatile("global_load_async_to_lds_b128 %0, %1, off\n\t"
                 "global_load_async_to_lds_b128 %0, %1, off offset:16"
                 :: "v"(lb), "v"(bP) : "memory");
    asm volatile("s_wait_asynccnt 0x0" ::: "memory");
  } else {
    v8h a0 = hz, a1 = hz;
    if (rowOK) { a0 = *(const v8h*)aP; a1 = *(const v8h*)(aP + 8); }
    *(v8h*)&As[sBase] = a0;
    *(v8h*)&As[sBase + 8] = a1;
    *(v8h*)&Bs[sBase] = *(const v8h*)bP;
    *(v8h*)&Bs[sBase + 8] = *(const v8h*)(bP + 8);
  }
  __syncthreads();

  int nsteps = K >> 5;
  for (int s = 0; s < nsteps; ++s) {
    int cur = (s & 1) * LDSL;
    int nxt = ((s + 1) & 1) * LDSL;
    bool have = (s + 1) < nsteps;
    // issue next tile's fills early; they overlap the WMMA block
    v8h na0 = hz, na1 = hz, nb0 = hz, nb1 = hz;
    if (have) {
      const _Float16* ap = aP + (size_t)(s + 1) * 32;
      const _Float16* bp = bP + (size_t)(s + 1) * 32;
      if (!GUARD) {
        unsigned la = (unsigned)(size_t)&As[nxt + sBase];
        unsigned lb = (unsigned)(size_t)&Bs[nxt + sBase];
        asm volatile("global_load_async_to_lds_b128 %0, %1, off\n\t"
                     "global_load_async_to_lds_b128 %0, %1, off offset:16"
                     :: "v"(la), "v"(ap) : "memory");
        asm volatile("global_load_async_to_lds_b128 %0, %1, off\n\t"
                     "global_load_async_to_lds_b128 %0, %1, off offset:16"
                     :: "v"(lb), "v"(bp) : "memory");
      } else {
        if (rowOK) { na0 = *(const v8h*)ap; na1 = *(const v8h*)(ap + 8); }
        nb0 = *(const v8h*)bp;
        nb1 = *(const v8h*)(bp + 8);
      }
      if (s + 2 < nsteps) {
        __builtin_prefetch(ap + 32, 0, 1);
        __builtin_prefetch(bp + 32, 0, 1);
      }
    }

    // fragments from current buffer
    v16h af[4], bf[2];
#pragma unroll
    for (int mt = 0; mt < 4; ++mt) {
      const _Float16* p = &As[cur + (waveM * 64 + mt * 16 + l15) * 40 + hs * 8];
      v8h lo = *(const v8h*)p, hi = *(const v8h*)(p + 16);
#pragma unroll
      for (int i = 0; i < 8; ++i) { af[mt][i] = lo[i]; af[mt][i + 8] = hi[i]; }
    }
#pragma unroll
    for (int nt = 0; nt < 2; ++nt) {
      const _Float16* p = &Bs[cur + (waveN * 32 + nt * 16 + l15) * 40 + hs * 8];
      v8h lo = *(const v8h*)p, hi = *(const v8h*)(p + 16);
#pragma unroll
      for (int i = 0; i < 8; ++i) { bf[nt][i] = lo[i]; bf[nt][i + 8] = hi[i]; }
    }
#pragma unroll
    for (int mt = 0; mt < 4; ++mt)
#pragma unroll
      for (int nt = 0; nt < 2; ++nt)
        acc[mt][nt] = __builtin_amdgcn_wmma_f32_16x16x32_f16(
            false, af[mt], false, bf[nt], (short)0, acc[mt][nt], false, false);

    // complete staging into the other buffer, one barrier per K-step
    if (have) {
      if (!GUARD) {
        asm volatile("s_wait_asynccnt 0x0" ::: "memory");
      } else {
        *(v8h*)&As[nxt + sBase] = na0;
        *(v8h*)&As[nxt + sBase + 8] = na1;
        *(v8h*)&Bs[nxt + sBase] = nb0;
        *(v8h*)&Bs[nxt + sBase + 8] = nb1;
      }
    }
    __syncthreads();
  }

  // epilogue: bias (+res) (gelu) -> f32 and/or f16
#pragma unroll
  for (int mt = 0; mt < 4; ++mt) {
    int rowLoc = waveM * 64 + mt * 16 + hs * 8;
#pragma unroll
    for (int nt = 0; nt < 2; ++nt) {
      size_t col = blockN + waveN * 32 + nt * 16 + l15;
      float bv = bias ? bias[col] : 0.f;
#pragma unroll
      for (int r = 0; r < 8; ++r) {
        size_t grow = blockM + rowLoc + r;
        if (!GUARD || grow < (size_t)M) {
          float v = acc[mt][nt][r] + bv;
          if (res) v += res[grow * N + col];
          if (do_gelu) v = 0.5f * v * (1.f + erff(v * 0.70710678f));
          if (C32) C32[grow * N + col] = v;
          if (C16) C16[grow * N + col] = (_Float16)v;
        }
      }
    }
  }
}

__global__ __launch_bounds__(256)
void gemm_kernel(const _Float16* __restrict__ A, const _Float16* __restrict__ Bt,
                 const float* __restrict__ bias, const float* __restrict__ res,
                 float* __restrict__ C32, _Float16* __restrict__ C16,
                 int M, int N, int K, int do_gelu) {
  __shared__ __align__(16) _Float16 As[2 * LDSL];
  __shared__ __align__(16) _Float16 Bs[2 * LDSL];
  if ((size_t)(blockIdx.y + 1) * 128 <= (size_t)M)
    gemm_body<false>(A, Bt, bias, res, C32, C16, M, N, K, do_gelu, As, Bs);
  else
    gemm_body<true>(A, Bt, bias, res, C32, C16, M, N, K, do_gelu, As, Bs);
}

// ---------------------------------------------------------------------------
// flash attention: per wave = 16 query rows, streams keys in chunks of 32
// Q,K: f16 [token][hid]; Vt: f16 [b][h][d][s]; out ctx f16 [token][hid]
__global__ __launch_bounds__(128)
void attn_kernel(const _Float16* __restrict__ Qh, const _Float16* __restrict__ Kh,
                 const _Float16* __restrict__ Vt, const float* __restrict__ mask,
                 _Float16* __restrict__ ctxh) {
  int lane = threadIdx.x & 31;
  int wid = threadIdx.x >> 5;
  int qt = blockIdx.x;  // 0..7
  int h  = blockIdx.y;  // head
  int b  = blockIdx.z;  // batch
  int qbase = qt * 64 + wid * 16;
  int l15 = lane & 15, hs = lane >> 4;

  __shared__ __align__(16) _Float16 pbuf[4][16][40];

  // Q A-fragments: d 0..31 and 32..63
  v16h aq[2];
  {
    const _Float16* qp = Qh + ((size_t)(b * SEQ + qbase + l15)) * HIDN + h * HDIM;
#pragma unroll
    for (int f = 0; f < 2; ++f) {
      const _Float16* base = qp + f * 32;
      v8h lo = *(const v8h*)(base + hs * 8);
      v8h hi = *(const v8h*)(base + 16 + hs * 8);
#pragma unroll
      for (int i = 0; i < 8; ++i) { aq[f][i] = lo[i]; aq[f][i + 8] = hi[i]; }
    }
  }

  v8f z = {};
  v8f ctx[4];
#pragma unroll
  for (int t = 0; t < 4; ++t) ctx[t] = z;
  float m[8], l[8];
#pragma unroll
  for (int r = 0; r < 8; ++r) { m[r] = -1e30f; l[r] = 0.f; }
  const float scale = 0.125f;  // 1/sqrt(64)

  for (int j = 0; j < SEQ; j += 32) {
    // scores 16x32 = two 16x16 C tiles, K-dim 64 = 2 wmma steps each
    v8f sc[2]; sc[0] = z; sc[1] = z;
#pragma unroll
    for (int t = 0; t < 2; ++t) {
      const _Float16* kp =
          Kh + ((size_t)(b * SEQ + j + t * 16 + l15)) * HIDN + h * HDIM;
#pragma unroll
      for (int f = 0; f < 2; ++f) {
        const _Float16* basep = kp + f * 32 + hs * 8;
        v8h lo = *(const v8h*)basep, hi = *(const v8h*)(basep + 16);
        v16h bk;
#pragma unroll
        for (int i = 0; i < 8; ++i) { bk[i] = lo[i]; bk[i + 8] = hi[i]; }
        sc[t] = __builtin_amdgcn_wmma_f32_16x16x32_f16(
            false, aq[f], false, bk, (short)0, sc[t], false, false);
      }
    }
    float mk0 = mask[b * SEQ + j + l15];
    float mk1 = mask[b * SEQ + j + 16 + l15];
#pragma unroll
    for (int r = 0; r < 8; ++r) {
      sc[0][r] = sc[0][r] * scale + mk0;
      sc[1][r] = sc[1][r] * scale + mk1;
    }
    // online softmax (row reductions across the 16-lane halves)
    float alpha[8];
#pragma unroll
    for (int r = 0; r < 8; ++r) {
      float cm = fmaxf(sc[0][r], sc[1][r]);
#pragma unroll
      for (int off = 1; off < 16; off <<= 1)
        cm = fmaxf(cm, __shfl_xor(cm, off, 32));
      float mn = fmaxf(m[r], cm);
      alpha[r] = __expf(m[r] - mn);
      m[r] = mn;
      sc[0][r] = __expf(sc[0][r] - mn);
      sc[1][r] = __expf(sc[1][r] - mn);
      float ps = sc[0][r] + sc[1][r];
#pragma unroll
      for (int off = 1; off < 16; off <<= 1) ps += __shfl_xor(ps, off, 32);
      l[r] = l[r] * alpha[r] + ps;
    }
#pragma unroll
    for (int t = 0; t < 4; ++t)
#pragma unroll
      for (int r = 0; r < 8; ++r) ctx[t][r] *= alpha[r];

    // P -> LDS (C layout) -> reload as A-fragment layout
#pragma unroll
    for (int t = 0; t < 2; ++t)
#pragma unroll
      for (int r = 0; r < 8; ++r)
        pbuf[wid][r + hs * 8][t * 16 + l15] = (_Float16)sc[t][r];

    v16h pf;
    {
      const _Float16* pp = &pbuf[wid][l15][hs * 8];
      v8h lo = *(const v8h*)pp, hi = *(const v8h*)(pp + 16);
#pragma unroll
      for (int i = 0; i < 8; ++i) { pf[i] = lo[i]; pf[i + 8] = hi[i]; }
    }

    // ctx += P @ V  (V via transposed layout: contiguous along keys)
#pragma unroll
    for (int dt = 0; dt < 4; ++dt) {
      const _Float16* vp =
          Vt + ((size_t)((b * NHEAD + h) * HDIM + dt * 16 + l15)) * SEQ + j + hs * 8;
      v8h lo = *(const v8h*)vp, hi = *(const v8h*)(vp + 16);
      v16h vf;
#pragma unroll
      for (int i = 0; i < 8; ++i) { vf[i] = lo[i]; vf[i + 8] = hi[i]; }
      ctx[dt] = __builtin_amdgcn_wmma_f32_16x16x32_f16(
          false, pf, false, vf, (short)0, ctx[dt], false, false);
    }
  }

  // normalize and write back to [token][hid] f16 for the output projection
#pragma unroll
  for (int dt = 0; dt < 4; ++dt)
#pragma unroll
    for (int r = 0; r < 8; ++r) {
      int row = qbase + r + hs * 8;
      float v = ctx[dt][r] / l[r];
      ctxh[((size_t)(b * SEQ + row)) * HIDN + h * HDIM + dt * 16 + l15] =
          (_Float16)v;
    }
}

// ---------------------------------------------------------------------------
// tiny dialog attention: 4 batches x 12 heads x 8 turns, mask (k>=q)->-1e4
__global__ void dlg_attn_kernel(const _Float16* __restrict__ Qh,
                                const _Float16* __restrict__ Kh,
                                const _Float16* __restrict__ Vh,
                                _Float16* __restrict__ out) {
  int tid = blockIdx.x * blockDim.x + threadIdx.x;
  if (tid >= 4 * NHEAD * 8) return;
  int q = tid & 7;
  int h = (tid >> 3) % NHEAD;
  int bp = tid / (NHEAD * 8);
  const _Float16* qp = Qh + (size_t)(bp * 8 + q) * HIDN + h * HDIM;
  float s[8];
  float mx = -1e30f;
  for (int k = 0; k < 8; ++k) {
    const _Float16* kp = Kh + (size_t)(bp * 8 + k) * HIDN + h * HDIM;
    float acc = 0.f;
    for (int d = 0; d < HDIM; ++d) acc += (float)qp[d] * (float)kp[d];
    acc = acc * 0.125f + ((k >= q) ? -10000.f : 0.f);
    s[k] = acc;
    mx = fmaxf(mx, acc);
  }
  float den = 0.f;
  for (int k = 0; k < 8; ++k) { s[k] = __expf(s[k] - mx); den += s[k]; }
  float inv = 1.f / den;
  _Float16* op = out + (size_t)(bp * 8 + q) * HIDN + h * HDIM;
  for (int d = 0; d < HDIM; ++d) {
    float acc = 0.f;
    for (int k = 0; k < 8; ++k)
      acc += s[k] * (float)Vh[(size_t)(bp * 8 + k) * HIDN + h * HDIM + d];
    op[d] = (_Float16)(acc * inv);
  }
}

// ---------------------------------------------------------------------------
extern "C" void kernel_launch(void* const* d_in, const int* in_sizes, int n_in,
                              void* d_out, int out_size, void* d_ws, size_t ws_size,
                              hipStream_t stream) {
  (void)in_sizes; (void)n_in; (void)out_size; (void)ws_size;
  const float* hidden = (const float*)d_in[0];
  const float* amask  = (const float*)d_in[1];
  const float* Wq  = (const float*)d_in[2];  const float* bq  = (const float*)d_in[3];
  const float* Wk  = (const float*)d_in[4];  const float* bk  = (const float*)d_in[5];
  const float* Wv  = (const float*)d_in[6];  const float* bv  = (const float*)d_in[7];
  const float* Wao = (const float*)d_in[8];  const float* bao = (const float*)d_in[9];
  const float* ln1g = (const float*)d_in[10]; const float* ln1b = (const float*)d_in[11];
  const float* dWq = (const float*)d_in[12]; const float* dbq = (const float*)d_in[13];
  const float* dWk = (const float*)d_in[14]; const float* dbk = (const float*)d_in[15];
  const float* dWv = (const float*)d_in[16]; const float* dbv = (const float*)d_in[17];
  const float* dWo = (const float*)d_in[18]; const float* dbo = (const float*)d_in[19];
  const float* dlng = (const float*)d_in[20]; const float* dlnb = (const float*)d_in[21];
  const float* Wi  = (const float*)d_in[22]; const float* bi  = (const float*)d_in[23];
  const float* Wo2 = (const float*)d_in[24]; const float* bo2 = (const float*)d_in[25];
  const float* ln2g = (const float*)d_in[26]; const float* ln2b = (const float*)d_in[27];

  char* p = (char*)d_ws;
  auto carve = [&](size_t bytes) -> void* {
    void* r = (void*)p;
    p += (bytes + 255) & ~(size_t)255;
    return r;
  };
  const size_t TH = (size_t)NTOK * HIDN;  // 12.58M elements
  _Float16* wtQ  = (_Float16*)carve((size_t)HIDN * HIDN * 2);
  _Float16* wtK  = (_Float16*)carve((size_t)HIDN * HIDN * 2);
  _Float16* wtV  = (_Float16*)carve((size_t)HIDN * HIDN * 2);
  _Float16* wtAO = (_Float16*)carve((size_t)HIDN * HIDN * 2);
  _Float16* wtdQ = (_Float16*)carve((size_t)HIDN * HIDN * 2);
  _Float16* wtdK = (_Float16*)carve((size_t)HIDN * HIDN * 2);
  _Float16* wtdV = (_Float16*)carve((size_t)HIDN * HIDN * 2);
  _Float16* wtdO = (_Float16*)carve((size_t)HIDN * HIDN * 2);
  _Float16* wtI  = (_Float16*)carve((size_t)HIDN * FFN * 2);
  _Float16* wtO2 = (_Float16*)carve((size_t)FFN * HIDN * 2);
  _Float16* xh   = (_Float16*)carve(TH * 2);
  _Float16* qh   = (_Float16*)carve(TH * 2);
  _Float16* kh   = (_Float16*)carve(TH * 2);
  _Float16* vh   = (_Float16*)carve(TH * 2);
  _Float16* vth  = (_Float16*)carve(TH * 2);
  _Float16* ctxh = (_Float16*)carve(TH * 2);
  float*    a1   = (float*)carve(TH * 4);     // attn-proj out; reused for FFN out
  float*    x1   = (float*)carve(TH * 4);
  _Float16* x1h  = (_Float16*)carve(TH * 2);
  _Float16* interh = (_Float16*)carve((size_t)NTOK * FFN * 2);
  _Float16* clsh  = (_Float16*)carve((size_t)32 * HIDN * 2);
  float*    clsf  = (float*)carve((size_t)32 * HIDN * 4);
  _Float16* dqh   = (_Float16*)carve((size_t)32 * HIDN * 2);
  _Float16* dkh   = (_Float16*)carve((size_t)32 * HIDN * 2);
  _Float16* dvh   = (_Float16*)carve((size_t)32 * HIDN * 2);
  _Float16* dctxh = (_Float16*)carve((size_t)32 * HIDN * 2);
  float*    da1   = (float*)carve((size_t)32 * HIDN * 4);
  float*    dx1   = (float*)carve((size_t)32 * HIDN * 4);
  _Float16* dx1h  = (_Float16*)carve((size_t)32 * HIDN * 2);

  // 1. weight transposes to f16 [N][K]
  int wElems = HIDN * HIDN, wGrid = (wElems + 255) / 256;
  transpose_w_kernel<<<wGrid, 256, 0, stream>>>(Wq,  wtQ,  HIDN, HIDN);
  transpose_w_kernel<<<wGrid, 256, 0, stream>>>(Wk,  wtK,  HIDN, HIDN);
  transpose_w_kernel<<<wGrid, 256, 0, stream>>>(Wv,  wtV,  HIDN, HIDN);
  transpose_w_kernel<<<wGrid, 256, 0, stream>>>(Wao, wtAO, HIDN, HIDN);
  transpose_w_kernel<<<wGrid, 256, 0, stream>>>(dWq, wtdQ, HIDN, HIDN);
  transpose_w_kernel<<<wGrid, 256, 0, stream>>>(dWk, wtdK, HIDN, HIDN);
  transpose_w_kernel<<<wGrid, 256, 0, stream>>>(dWv, wtdV, HIDN, HIDN);
  transpose_w_kernel<<<wGrid, 256, 0, stream>>>(dWo, wtdO, HIDN, HIDN);
  int wiElems = HIDN * FFN;
  transpose_w_kernel<<<(wiElems + 255) / 256, 256, 0, stream>>>(Wi, wtI, HIDN, FFN);
  transpose_w_kernel<<<(wiElems + 255) / 256, 256, 0, stream>>>(Wo2, wtO2, FFN, HIDN);

  // 2. activations to f16
  cvt_f32_f16_kernel<<<(int)((TH + 255) / 256), 256, 0, stream>>>(hidden, xh, (int)TH);

  // 3. QKV projections (f16 outputs only)
  dim3 gproj((HIDN + 127) / 128, (NTOK + 127) / 128);
  gemm_kernel<<<gproj, 256, 0, stream>>>(xh, wtQ, bq, nullptr, nullptr, qh, NTOK, HIDN, HIDN, 0);
  gemm_kernel<<<gproj, 256, 0, stream>>>(xh, wtK, bk, nullptr, nullptr, kh, NTOK, HIDN, HIDN, 0);
  gemm_kernel<<<gproj, 256, 0, stream>>>(xh, wtV, bv, nullptr, nullptr, vh, NTOK, HIDN, HIDN, 0);
  transpose_v_kernel<<<(int)((TH + 255) / 256), 256, 0, stream>>>(vh, vth);

  // 4. flash attention
  dim3 gattn(SEQ / 64, NHEAD, BATCH);
  attn_kernel<<<gattn, 128, 0, stream>>>(qh, kh, vth, amask, ctxh);

  // 5. output projection + residual, then LN1
  gemm_kernel<<<gproj, 256, 0, stream>>>(ctxh, wtAO, bao, hidden, a1, nullptr, NTOK, HIDN, HIDN, 0);
  ln_kernel<<<NTOK, 256, 0, stream>>>(a1, ln1g, ln1b, x1, x1h);

  // 6. dialog block on CLS tokens (32 rows)
  int cGrid = (32 * HIDN + 255) / 256;
  gather_cls_kernel<<<cGrid, 256, 0, stream>>>(x1, x1h, clsf, clsh);
  dim3 gdlg((HIDN + 127) / 128, 1);
  gemm_kernel<<<gdlg, 256, 0, stream>>>(clsh, wtdQ, dbq, nullptr, nullptr, dqh, 32, HIDN, HIDN, 0);
  gemm_kernel<<<gdlg, 256, 0, stream>>>(clsh, wtdK, dbk, nullptr, nullptr, dkh, 32, HIDN, HIDN, 0);
  gemm_kernel<<<gdlg, 256, 0, stream>>>(clsh, wtdV, dbv, nullptr, nullptr, dvh, 32, HIDN, HIDN, 0);
  dlg_attn_kernel<<<2, 256, 0, stream>>>(dqh, dkh, dvh, dctxh);
  gemm_kernel<<<gdlg, 256, 0, stream>>>(dctxh, wtdO, dbo, clsf, da1, nullptr, 32, HIDN, HIDN, 0);
  ln_kernel<<<32, 256, 0, stream>>>(da1, dlng, dlnb, dx1, dx1h);
  scatter_cls_kernel<<<cGrid, 256, 0, stream>>>(dx1, dx1h, x1, x1h);

  // 7. FFN: inter = gelu(x1 @ Wi + bi) (f16 only), out = inter @ Wo2 + bo2 + x1
  dim3 gffn1((FFN + 127) / 128, (NTOK + 127) / 128);
  gemm_kernel<<<gffn1, 256, 0, stream>>>(x1h, wtI, bi, nullptr, nullptr, interh, NTOK, FFN, HIDN, 1);
  gemm_kernel<<<gproj, 256, 0, stream>>>(interh, wtO2, bo2, x1, a1, nullptr, NTOK, HIDN, FFN, 0);

  // 8. final LayerNorm -> d_out
  ln_kernel<<<NTOK, 256, 0, stream>>>(a1, ln2g, ln2b, (float*)d_out, nullptr);
}
